// RelTMCell_25391846654697
// MI455X (gfx1250) — matrix operand order repaired and checked by reading, not compiled, not run on hardware
//
#include <hip/hip_runtime.h>
#include <hip/hip_bf16.h>
#include <math.h>

#define HD   128
#define NH   4
#define DH   32
#define ZD   256
#define LNEPS 1e-5f

typedef __attribute__((ext_vector_type(16))) _Float16 v16h;
typedef __attribute__((ext_vector_type(8)))  _Float16 v8h;
typedef __attribute__((ext_vector_type(4)))  _Float16 v4h;
typedef __attribute__((ext_vector_type(8)))  float    v8f;

static __device__ inline v8f zero8() {
    v8f z;
#pragma unroll
    for (int i = 0; i < 8; ++i) z[i] = 0.0f;
    return z;
}

// A fragment: 16(M) x 32(K) fp16, row-major source with stride ld (halves).
// ISA 7.12.2: lane m=l&15, sel=l>>4; halves 0..7 = K (sel*8 .. sel*8+7),
// halves 8..15 = K (16+sel*8 .. 16+sel*8+7)  (relative to k0).
static __device__ inline v16h load_frag_a(const _Float16* base, int ld, int k0, int lane) {
    int m = lane & 15, sel = lane >> 4;
    const _Float16* p0 = base + m * ld + k0 + sel * 8;
    v8h lo = *(const v8h*)p0;
    v8h hi = *(const v8h*)(p0 + 16);
    v16h a;
#pragma unroll
    for (int i = 0; i < 8; ++i) { a[i] = lo[i]; a[i + 8] = hi[i]; }
    return a;
}

// B fragment: 32(K) x 16(N) fp16 from transposed weight WT (row n, K contiguous,
// stride ld). lane n=l&15, sel=l>>4; halves 0..7 = K (sel*8 .. sel*8+7), halves 8..15 = K (16+sel*8 .. 16+sel*8+7).
static __device__ inline v16h load_frag_b(const _Float16* bt, int ld, int n0, int k0, int lane) {
    int n = n0 + (lane & 15), sel = lane >> 4;
    const _Float16* p = bt + n * ld + k0 + sel * 8;
    v8h lo = *(const v8h*)p;
    v8h hi = *(const v8h*)(p + 16);
    v16h b;
#pragma unroll
    for (int i = 0; i < 8; ++i) { b[i] = lo[i]; b[i + 8] = hi[i]; }
    return b;
}

static __device__ inline v8f wmma16(v16h a, v16h b, v8f c) {
    return __builtin_amdgcn_wmma_f32_16x16x32_f16(false, a, false, b, (short)0, c, false, false);
}

// ---------------- prep kernels ----------------

// WT[n*K + k] = W[k*Nn + n] in fp16.  W is (K x Nn) row-major.
__global__ void k_transpose16(const float* __restrict__ W, _Float16* __restrict__ WT, int K, int Nn) {
    int idx = blockIdx.x * blockDim.x + threadIdx.x;
    if (idx < K * Nn) {
        int n = idx / K, k = idx - n * K;
        WT[idx] = (_Float16)W[k * Nn + n];
    }
}

__global__ void k_cvt16(const float* __restrict__ x, _Float16* __restrict__ y, int n) {
    int idx = blockIdx.x * blockDim.x + threadIdx.x;
    if (idx < n) y[idx] = (_Float16)x[idx];
}

// CSR offsets from sorted dst: offs[n] = lower_bound(dst, n), offs[N] = E.
__global__ void k_row_offsets(const int* __restrict__ dst, int* __restrict__ offs, int N, int E) {
    int n = blockIdx.x * blockDim.x + threadIdx.x;
    if (n > N) return;
    int lo = 0, hi = E;
    while (lo < hi) { int mid = (lo + hi) >> 1; if (dst[mid] < n) lo = mid + 1; else hi = mid; }
    offs[n] = lo;
}

// ---------------- q = h @ Wq ----------------
__global__ void k_qgemm(const _Float16* __restrict__ h16, const _Float16* __restrict__ WqT,
                        float* __restrict__ q) {
    int lane = threadIdx.x & 31, w = threadIdx.x >> 5;
    int row0 = blockIdx.x * 16;
    const _Float16* A = h16 + row0 * HD;
    v8f c0 = zero8(), c1 = zero8();
#pragma unroll
    for (int kb = 0; kb < 4; ++kb) {
        int k0 = kb * 32;
        v16h a  = load_frag_a(A, HD, k0, lane);
        v16h b0 = load_frag_b(WqT, HD, w * 32, k0, lane);
        v16h b1 = load_frag_b(WqT, HD, w * 32 + 16, k0, lane);
        c0 = wmma16(a, b0, c0);
        c1 = wmma16(a, b1, c1);
    }
    int m0 = (lane >> 4) * 8, n = lane & 15;
#pragma unroll
    for (int r = 0; r < 8; ++r) {
        q[(row0 + m0 + r) * HD + w * 32 + n]      = c0[r];
        q[(row0 + m0 + r) * HD + w * 32 + 16 + n] = c1[r];
    }
}

// ---------------- fused edge kernel: hs, k, v, logits ----------------
__global__ void k_edges(const _Float16* __restrict__ h16, const float* __restrict__ relv,
                        const int* __restrict__ src, const int* __restrict__ relid,
                        const int* __restrict__ dst,
                        const _Float16* __restrict__ WkT, const _Float16* __restrict__ WvT,
                        const float* __restrict__ q,
                        float* __restrict__ logits, _Float16* __restrict__ v16) {
    __shared__ alignas(16) _Float16 hsT[16 * HD];   // hs tile fp16
    __shared__ alignas(16) float    kT[16 * HD];    // k tile fp32
    __shared__ alignas(16) float    vT[16 * HD];    // v tile fp32
    int t = threadIdx.x, lane = t & 31, w = t >> 5;
    int e0 = blockIdx.x * 16;

    // phase 1: hs = lrelu(h16[src] + relvec, 0.25) -> fp16 LDS (vector loads)
    {
        int row = t >> 3, seg = t & 7;
        int e = e0 + row;
        int s = src[e], rl = relid[e];
        const v8h* hp = (const v8h*)(h16 + (size_t)s * HD + seg * 16);
        const float* rp = relv + rl * HD + seg * 16;
        v8h a0 = hp[0], a1 = hp[1];
        float rb[16];
        *(float4*)&rb[0]  = *(const float4*)(rp + 0);
        *(float4*)&rb[4]  = *(const float4*)(rp + 4);
        *(float4*)&rb[8]  = *(const float4*)(rp + 8);
        *(float4*)&rb[12] = *(const float4*)(rp + 12);
        v8h o0, o1;
#pragma unroll
        for (int i = 0; i < 8; ++i) {
            float x0 = (float)a0[i] + rb[i];
            float x1 = (float)a1[i] + rb[8 + i];
            o0[i] = (_Float16)(x0 > 0.0f ? x0 : 0.25f * x0);
            o1[i] = (_Float16)(x1 > 0.0f ? x1 : 0.25f * x1);
        }
        *(v8h*)&hsT[row * HD + seg * 16]     = o0;
        *(v8h*)&hsT[row * HD + seg * 16 + 8] = o1;
    }
    __syncthreads();

    // phase 2: wave w computes head-w cols of k and v via WMMA
    {
        v8f c0 = zero8(), c1 = zero8(), d0 = zero8(), d1 = zero8();
#pragma unroll
        for (int kb = 0; kb < 4; ++kb) {
            int k0 = kb * 32;
            v16h a   = load_frag_a(hsT, HD, k0, lane);
            v16h bk0 = load_frag_b(WkT, HD, w * 32, k0, lane);
            v16h bk1 = load_frag_b(WkT, HD, w * 32 + 16, k0, lane);
            v16h bv0 = load_frag_b(WvT, HD, w * 32, k0, lane);
            v16h bv1 = load_frag_b(WvT, HD, w * 32 + 16, k0, lane);
            c0 = wmma16(a, bk0, c0);
            c1 = wmma16(a, bk1, c1);
            d0 = wmma16(a, bv0, d0);
            d1 = wmma16(a, bv1, d1);
        }
        int m0 = (lane >> 4) * 8, n = lane & 15;
#pragma unroll
        for (int r = 0; r < 8; ++r) {
            kT[(m0 + r) * HD + w * 32 + n]      = c0[r];
            kT[(m0 + r) * HD + w * 32 + 16 + n] = c1[r];
            vT[(m0 + r) * HD + w * 32 + n]      = d0[r];
            vT[(m0 + r) * HD + w * 32 + 16 + n] = d1[r];
        }
    }
    __syncthreads();

    // phase 3: logits[e][h] = <q[dst[e], head h], k[e, head h]> / sqrt(dh)
    if (t < 64) {
        int row = t & 15, hh = t >> 4;
        int e = e0 + row;
        const float* qp = q + dst[e] * HD + hh * DH;
        float s = 0.0f;
#pragma unroll
        for (int j = 0; j < DH; ++j) s += kT[row * HD + hh * DH + j] * qp[j];
        logits[e * NH + hh] = s * 0.17677669529663687f;  // 1/sqrt(32)
    }

    // phase 4: spill v tile as fp16 (non-temporal: 205MB stream, keep L2 for q/weights)
    {
        int row = t >> 3, seg = t & 7;
        const float* vrow = &vT[row * HD + seg * 16];
        v8h o0, o1;
#pragma unroll
        for (int i = 0; i < 8; ++i) { o0[i] = (_Float16)vrow[i]; o1[i] = (_Float16)vrow[8 + i]; }
        _Float16* dp = v16 + (size_t)(e0 + row) * HD + seg * 16;
        __builtin_nontemporal_store(o0, (v8h*)dp);
        __builtin_nontemporal_store(o1, (v8h*)(dp + 8));
    }
}

// ---------------- segment softmax + weighted v reduction ----------------
__global__ void k_reduce(const int* __restrict__ offs, const float* __restrict__ logits,
                         const _Float16* __restrict__ v16, float* __restrict__ red, int N) {
    int node = blockIdx.x * 4 + (threadIdx.x >> 5);
    int lane = threadIdx.x & 31;
    if (node >= N) return;
    int beg = offs[node], end = offs[node + 1];

    if (beg == end) {  // isolated node: red row = 0
        float4 z = {0.f, 0.f, 0.f, 0.f};
        *(float4*)(red + node * HD + lane * 4) = z;
        return;
    }

    float mx[NH];
#pragma unroll
    for (int h = 0; h < NH; ++h) mx[h] = -3.4e38f;
    for (int e = beg + lane; e < end; e += 32)
#pragma unroll
        for (int h = 0; h < NH; ++h) mx[h] = fmaxf(mx[h], logits[e * NH + h]);
#pragma unroll
    for (int h = 0; h < NH; ++h)
        for (int off = 16; off > 0; off >>= 1)
            mx[h] = fmaxf(mx[h], __shfl_xor(mx[h], off, 32));

    float sm[NH];
#pragma unroll
    for (int h = 0; h < NH; ++h) sm[h] = 0.0f;
    for (int e = beg + lane; e < end; e += 32)
#pragma unroll
        for (int h = 0; h < NH; ++h) sm[h] += expf(logits[e * NH + h] - mx[h]);
#pragma unroll
    for (int h = 0; h < NH; ++h)
        for (int off = 16; off > 0; off >>= 1)
            sm[h] += __shfl_xor(sm[h], off, 32);

    int myh = lane >> 3;                 // cols [lane*4, lane*4+4) all in head lane/8
    float m_h = mx[myh], rd_h = 1.0f / sm[myh];

    float acc[4] = {0.f, 0.f, 0.f, 0.f};
    for (int e = beg; e < end; ++e) {
        float a = expf(logits[e * NH + myh] - m_h) * rd_h;
        v4h vv = __builtin_nontemporal_load((const v4h*)(v16 + (size_t)e * HD + lane * 4));
#pragma unroll
        for (int i = 0; i < 4; ++i) acc[i] += a * (float)vv[i];
    }
    float4 o = {acc[0], acc[1], acc[2], acc[3]};
    *(float4*)(red + node * HD + lane * 4) = o;
}

// ---------------- fused node kernel: Wal + LN1 + FFN + LN2 ----------------
__global__ void k_node(const float* __restrict__ red, const float* __restrict__ h,
                       const _Float16* __restrict__ WalT, const float* __restrict__ bal,
                       const float* __restrict__ g1, const float* __restrict__ b1,
                       const _Float16* __restrict__ W1T, const float* __restrict__ bb1,
                       const _Float16* __restrict__ W2T, const float* __restrict__ bb2,
                       const float* __restrict__ g2, const float* __restrict__ b2,
                       float* __restrict__ out) {
    __shared__ alignas(16) _Float16 ah[16 * HD];   // fp16 A tile (red, then h1)
    __shared__ alignas(16) _Float16 az[16 * ZD];   // fp16 z tile (celu output)
    __shared__ alignas(16) float    fb[16 * HD];   // fp32 staging (summ, h1, h1+x)
    int t = threadIdx.x, lane = t & 31, w = t >> 5;
    int row0 = blockIdx.x * 16;
    int m0 = (lane >> 4) * 8, nn = lane & 15;

    // phase 1: ah <- fp16(red tile)
    {
        int row = t >> 3, seg = t & 7;
        const float* rp = red + (size_t)(row0 + row) * HD + seg * 16;
        float rb[16];
        *(float4*)&rb[0]  = *(const float4*)(rp + 0);
        *(float4*)&rb[4]  = *(const float4*)(rp + 4);
        *(float4*)&rb[8]  = *(const float4*)(rp + 8);
        *(float4*)&rb[12] = *(const float4*)(rp + 12);
        v8h o0, o1;
#pragma unroll
        for (int i = 0; i < 8; ++i) { o0[i] = (_Float16)rb[i]; o1[i] = (_Float16)rb[8 + i]; }
        *(v8h*)&ah[row * HD + seg * 16]     = o0;
        *(v8h*)&ah[row * HD + seg * 16 + 8] = o1;
    }
    __syncthreads();

    // phase 2: summ = red @ Wal + bal
    {
        v8f c0 = zero8(), c1 = zero8();
#pragma unroll
        for (int kb = 0; kb < 4; ++kb) {
            int k0 = kb * 32;
            v16h a  = load_frag_a(ah, HD, k0, lane);
            v16h b0 = load_frag_b(WalT, HD, w * 32, k0, lane);
            v16h b1 = load_frag_b(WalT, HD, w * 32 + 16, k0, lane);
            c0 = wmma16(a, b0, c0);
            c1 = wmma16(a, b1, c1);
        }
        float ba0 = bal[w * 32 + nn], ba1 = bal[w * 32 + 16 + nn];
#pragma unroll
        for (int r = 0; r < 8; ++r) {
            fb[(m0 + r) * HD + w * 32 + nn]      = c0[r] + ba0;
            fb[(m0 + r) * HD + w * 32 + 16 + nn] = c1[r] + ba1;
        }
    }
    __syncthreads();

    // phase 3: h1 = LN(summ + h): 8 lanes per row, shfl reduction
    {
        int row = t >> 3, seg = t & 7, j0 = seg * 16;
        int grow = row0 + row;
        const float* hp = h + (size_t)grow * HD + j0;
        float x[16];
        float sum = 0.f, sq = 0.f;
#pragma unroll
        for (int i = 0; i < 16; ++i) {
            x[i] = fb[row * HD + j0 + i] + hp[i];
            sum += x[i]; sq += x[i] * x[i];
        }
#pragma unroll
        for (int off = 1; off < 8; off <<= 1) {
            sum += __shfl_xor(sum, off, 32);
            sq  += __shfl_xor(sq, off, 32);
        }
        float mu = sum * (1.0f / HD);
        float var = sq * (1.0f / HD) - mu * mu;
        float rs = rsqrtf(var + LNEPS);
#pragma unroll
        for (int i = 0; i < 16; ++i) {
            float y = (x[i] - mu) * rs * g1[j0 + i] + b1[j0 + i];
            fb[row * HD + j0 + i] = y;
            ah[row * HD + j0 + i] = (_Float16)y;
        }
    }
    __syncthreads();

    // phase 4: z = celu(h1 @ W1 + bb1) -> fp16 az   (wave w: cols [64w, 64w+64))
    {
        v8f cc[4];
#pragma unroll
        for (int tt = 0; tt < 4; ++tt) cc[tt] = zero8();
        int nb = w * 64;
#pragma unroll
        for (int kb = 0; kb < 4; ++kb) {
            int k0 = kb * 32;
            v16h a = load_frag_a(ah, HD, k0, lane);
#pragma unroll
            for (int tt = 0; tt < 4; ++tt) {
                v16h b = load_frag_b(W1T, HD, nb + tt * 16, k0, lane);
                cc[tt] = wmma16(a, b, cc[tt]);
            }
        }
#pragma unroll
        for (int tt = 0; tt < 4; ++tt) {
            int col = nb + tt * 16 + nn;
            float bbv = bb1[col];
#pragma unroll
            for (int r = 0; r < 8; ++r) {
                float val = cc[tt][r] + bbv;
                val = val > 0.0f ? val : (expf(val) - 1.0f);   // celu, alpha=1
                az[(m0 + r) * ZD + col] = (_Float16)val;
            }
        }
    }
    __syncthreads();

    // phase 5: x = z @ W2 + bb2; fb <- h1 + x
    {
        v8f c0 = zero8(), c1 = zero8();
#pragma unroll
        for (int kb = 0; kb < 8; ++kb) {
            int k0 = kb * 32;
            v16h a  = load_frag_a(az, ZD, k0, lane);
            v16h b0 = load_frag_b(W2T, ZD, w * 32, k0, lane);
            v16h b1 = load_frag_b(W2T, ZD, w * 32 + 16, k0, lane);
            c0 = wmma16(a, b0, c0);
            c1 = wmma16(a, b1, c1);
        }
        float bb0 = bb2[w * 32 + nn], bb1v = bb2[w * 32 + 16 + nn];
#pragma unroll
        for (int r = 0; r < 8; ++r) {
            int i0 = (m0 + r) * HD + w * 32 + nn;
            int i1 = (m0 + r) * HD + w * 32 + 16 + nn;
            fb[i0] = fb[i0] + c0[r] + bb0;
            fb[i1] = fb[i1] + c1[r] + bb1v;
        }
    }
    __syncthreads();

    // phase 6: out = LN(h1 + x): 8 lanes per row, shfl reduction
    {
        int row = t >> 3, seg = t & 7, j0 = seg * 16;
        int grow = row0 + row;
        float x[16];
        float sum = 0.f, sq = 0.f;
#pragma unroll
        for (int i = 0; i < 16; ++i) {
            x[i] = fb[row * HD + j0 + i];
            sum += x[i]; sq += x[i] * x[i];
        }
#pragma unroll
        for (int off = 1; off < 8; off <<= 1) {
            sum += __shfl_xor(sum, off, 32);
            sq  += __shfl_xor(sq, off, 32);
        }
        float mu = sum * (1.0f / HD);
        float var = sq * (1.0f / HD) - mu * mu;
        float rs = rsqrtf(var + LNEPS);
        float* op = out + (size_t)grow * HD + j0;
#pragma unroll
        for (int i = 0; i < 16; ++i)
            op[i] = (x[i] - mu) * rs * g2[j0 + i] + b2[j0 + i];
    }
}

// ---------------- launch ----------------
extern "C" void kernel_launch(void* const* d_in, const int* in_sizes, int n_in,
                              void* d_out, int out_size, void* d_ws, size_t ws_size,
                              hipStream_t stream) {
    const float* h    = (const float*)d_in[0];
    const int*   src  = (const int*)d_in[1];
    const int*   dst  = (const int*)d_in[2];
    const int*   rel  = (const int*)d_in[3];
    const float* relv = (const float*)d_in[4];
    const float* Wq   = (const float*)d_in[5];
    const float* Wk   = (const float*)d_in[6];
    const float* Wv   = (const float*)d_in[7];
    const float* Wal  = (const float*)d_in[8];
    const float* bal  = (const float*)d_in[9];
    const float* g1   = (const float*)d_in[10];
    const float* b1   = (const float*)d_in[11];
    const float* W1   = (const float*)d_in[12];
    const float* bb1  = (const float*)d_in[13];
    const float* W2   = (const float*)d_in[14];
    const float* bb2  = (const float*)d_in[15];
    const float* g2   = (const float*)d_in[16];
    const float* b2   = (const float*)d_in[17];
    float* out = (float*)d_out;

    const int N = in_sizes[0] / HD;
    const int E = in_sizes[1];

    // workspace carve-out (256B aligned)
    char* base = (char*)d_ws;
    size_t off = 0;
    auto alloc = [&](size_t bytes) -> void* {
        void* p = base + off;
        off = (off + bytes + 255) & ~(size_t)255;
        return p;
    };
    int*       offs = (int*)alloc((size_t)(N + 1) * 4);
    _Float16*  h16  = (_Float16*)alloc((size_t)N * HD * 2);
    _Float16*  WqT  = (_Float16*)alloc((size_t)HD * HD * 2);
    _Float16*  WkT  = (_Float16*)alloc((size_t)HD * HD * 2);
    _Float16*  WvT  = (_Float16*)alloc((size_t)HD * HD * 2);
    _Float16*  WalT = (_Float16*)alloc((size_t)HD * HD * 2);
    _Float16*  W1T  = (_Float16*)alloc((size_t)HD * ZD * 2);
    _Float16*  W2T  = (_Float16*)alloc((size_t)ZD * HD * 2);
    float*     q    = (float*)alloc((size_t)N * HD * 4);
    float*     lg   = (float*)alloc((size_t)E * NH * 4);
    _Float16*  v16  = (_Float16*)alloc((size_t)E * HD * 2);
    float*     red  = (float*)alloc((size_t)N * HD * 4);

    // prep
    k_transpose16<<<(HD * HD + 255) / 256, 256, 0, stream>>>(Wq,  WqT,  HD, HD);
    k_transpose16<<<(HD * HD + 255) / 256, 256, 0, stream>>>(Wk,  WkT,  HD, HD);
    k_transpose16<<<(HD * HD + 255) / 256, 256, 0, stream>>>(Wv,  WvT,  HD, HD);
    k_transpose16<<<(HD * HD + 255) / 256, 256, 0, stream>>>(Wal, WalT, HD, HD);
    k_transpose16<<<(HD * ZD + 255) / 256, 256, 0, stream>>>(W1,  W1T,  HD, ZD);
    k_transpose16<<<(ZD * HD + 255) / 256, 256, 0, stream>>>(W2,  W2T,  ZD, HD);
    k_cvt16<<<(N * HD + 255) / 256, 256, 0, stream>>>(h, h16, N * HD);
    k_row_offsets<<<(N + 1 + 255) / 256, 256, 0, stream>>>(dst, offs, N, E);

    // pipeline
    k_qgemm<<<N / 16, 128, 0, stream>>>(h16, WqT, q);
    k_edges<<<E / 16, 128, 0, stream>>>(h16, relv, src, rel, dst, WkT, WvT, q, lg, v16);
    k_reduce<<<(N + 3) / 4, 128, 0, stream>>>(offs, lg, v16, red, N);
    k_node<<<N / 16, 128, 0, stream>>>(red, h, WalT, bal, g1, b1, W1T, bb1, W2T, bb2, g2, b2, out);
}
